// HeteroGNN_81896436400720
// MI455X (gfx1250) — matrix operand order repaired and checked by reading, not compiled, not run on hardware
//
#include <hip/hip_runtime.h>
#include <hip/hip_bf16.h>

typedef __attribute__((ext_vector_type(16))) __bf16 v16bf;
typedef __attribute__((ext_vector_type(8)))  __bf16 v8bf;
typedef __attribute__((ext_vector_type(8)))  float  v8f;

#define NP   100000
#define NG   100000
#define NPA  20000
#define NEDGE 500000
#define INF  480
#define HD   128

__device__ __forceinline__ __bf16 f2bf(float f) { return (__bf16)f; }

__device__ __forceinline__ v16bf cat8(v8bf lo, v8bf hi) {
    return __builtin_shufflevector(lo, hi, 0, 1, 2, 3, 4, 5, 6, 7,
                                   8, 9, 10, 11, 12, 13, 14, 15);
}

// ---------------------------------------------------------------- utilities
__global__ void zero_kernel(unsigned* __restrict__ p, size_t n) {
    size_t i = (size_t)blockIdx.x * blockDim.x + threadIdx.x;
    size_t stride = (size_t)gridDim.x * blockDim.x;
    for (; i < n; i += stride) p[i] = 0u;
}

__global__ void degree_kernel(const int* __restrict__ dst, int e, int* __restrict__ cnt) {
    int i = blockIdx.x * blockDim.x + threadIdx.x;
    if (i < e) atomicAdd(&cnt[dst[i]], 1);
}

__global__ void inv_kernel(const int* __restrict__ cnt, float* __restrict__ inv, int n) {
    int i = blockIdx.x * blockDim.x + threadIdx.x;
    if (i < n) { int c = cnt[i]; inv[i] = 1.0f / (float)(c > 0 ? c : 1); }
}

// effective bias per (layer, dst-type): mean of bl over relations targeting dst
__global__ void beff_kernel(const float* __restrict__ bl, float* __restrict__ beff) {
    int i = threadIdx.x;          // 0..127
    int idx = blockIdx.x;         // 0..5 : l*3 + dt
    int l = idx / 3, t = idx % 3;
    float v;
    if (t == 0)      v = 0.5f * (bl[(l * 4 + 0) * HD + i] + bl[(l * 4 + 2) * HD + i]);
    else if (t == 1) v = bl[(l * 4 + 1) * HD + i];
    else             v = bl[(l * 4 + 3) * HD + i];
    beff[idx * HD + i] = v;
}

// ------------------------------------------------- weight -> WMMA B-fragment
// B layout (16-bit B 32x16): lane L holds column n = L%16, element e holds
// K = 8*(L/16) + e (e<8), 16 + 8*(L/16) + (e-8) (e>=8) within the 32-K tile. Packed as [frag][lane][16] bf16 so
// the GEMM loads one contiguous v16bf per lane.
__global__ void pack_frags(const float* __restrict__ W0, const float* __restrict__ W1,
                           float s, __bf16* __restrict__ out) {
    const int frag = blockIdx.x;            // kt*8 + nt
    const int kt = frag >> 3, nt = frag & 7;
    const int lane = threadIdx.x;
    const int lm = lane & 15, lhi = lane >> 4;
    const int n = nt * 16 + lm;
    __bf16* o = out + (size_t)frag * 512 + lane * 16;
#pragma unroll
    for (int e = 0; e < 16; ++e) {
        int k = kt * 32 + 8 * lhi + (e < 8 ? e : e + 8);
        float v = W0[(size_t)k * HD + n];
        if (W1) v += W1[(size_t)k * HD + n];
        o[e] = f2bf(v * s);
    }
}

// -------------------------------------------------------- edge scatter (sum)
__global__ void scatter_kernel(const float* __restrict__ Hsrc, const int* __restrict__ src,
                               const int* __restrict__ dst, float* __restrict__ agg, int e) {
    long idx = (long)blockIdx.x * blockDim.x + threadIdx.x;   // E*32 threads
    if (idx >= (long)e * 32) return;
    int ed = (int)(idx >> 5);
    int c  = (int)(idx & 31) << 2;
    int s = src[ed], d = dst[ed];
    const float4 v = *(const float4*)(Hsrc + (size_t)s * HD + c);
    float* a = agg + (size_t)d * HD + c;
    atomicAdd(a + 0, v.x); atomicAdd(a + 1, v.y);
    atomicAdd(a + 2, v.z); atomicAdd(a + 3, v.w);
}

// ------------------------------------------------- input projection (K=480)
// 32 rows/block; bf16 A tile staged in LDS; each B fragment reused by 2 WMMAs.
__global__ void __launch_bounds__(256)
proj_wmma(const float* __restrict__ X, const __bf16* __restrict__ wfrag,
          const float* __restrict__ bias, float* __restrict__ Y) {
    __shared__ __bf16 Xbf[32 * INF];
    const int tid = threadIdx.x;
    const int row0 = blockIdx.x * 32;

    const float4* s4 = (const float4*)(X + (size_t)row0 * INF);
    for (int i = tid; i < (32 * INF) / 4; i += 256) {
        float4 v = s4[i];
        __bf16* o = Xbf + i * 4;
        o[0] = f2bf(v.x); o[1] = f2bf(v.y); o[2] = f2bf(v.z); o[3] = f2bf(v.w);
    }
    if (tid == 0 && blockIdx.x + 1 < gridDim.x)
        __builtin_prefetch(X + (size_t)(row0 + 32) * INF, 0, 1);
    __syncthreads();

    const int wave = tid >> 5, lane = tid & 31;
    const int lm = lane & 15, lhi = lane >> 4;

    float bv = bias[wave * 16 + lm];
    v8f acc0, acc1;
#pragma unroll
    for (int v = 0; v < 8; ++v) { acc0[v] = bv; acc1[v] = bv; }

    const v16bf* bf = (const v16bf*)wfrag;
    for (int kt = 0; kt < INF / 32; ++kt) {
        v16bf b = bf[(size_t)(kt * 8 + wave) * 32 + lane];
        const int kb = kt * 32 + 8 * lhi;
        v16bf a0 = cat8(*(const v8bf*)(Xbf + lm * INF + kb),
                        *(const v8bf*)(Xbf + lm * INF + kb + 16));
        acc0 = __builtin_amdgcn_wmma_f32_16x16x32_bf16(false, a0, false, b,
                                                       (short)0, acc0, false, false);
        v16bf a1 = cat8(*(const v8bf*)(Xbf + (16 + lm) * INF + kb),
                        *(const v8bf*)(Xbf + (16 + lm) * INF + kb + 16));
        acc1 = __builtin_amdgcn_wmma_f32_16x16x32_bf16(false, a1, false, b,
                                                       (short)0, acc1, false, false);
    }
#pragma unroll
    for (int v = 0; v < 8; ++v) {
        float r0 = acc0[v];
        float r1 = acc1[v];
        Y[(size_t)(row0 + v + 8 * lhi) * HD + wave * 16 + lm]      = r0 > 0.f ? r0 : 0.f;
        Y[(size_t)(row0 + 16 + v + 8 * lhi) * HD + wave * 16 + lm] = r1 > 0.f ? r1 : 0.f;
    }
}

// ------------------------------------ 128x128 GEMM: C (+)= (A * rowscale) @ W
// 32 rows/block, B fragments cached in registers across both row tiles.
// accumulate==0: acc initialized from bias; accumulate==1: from C.
// relu: applied on final store.
__global__ void __launch_bounds__(256)
gemm128_wmma(const float* __restrict__ A, const float* __restrict__ rowscale,
             const __bf16* __restrict__ wfrag, const float* __restrict__ bias,
             float* __restrict__ C, int accumulate, int relu) {
    __shared__ __bf16 Abf[32 * HD];
    const int tid = threadIdx.x;
    const int row0 = blockIdx.x * 32;

    // cooperative load + row-scale + f32->bf16 convert into LDS (once/block)
    const float4* s4 = (const float4*)(A + (size_t)row0 * HD);
    for (int i = tid; i < (32 * HD) / 4; i += 256) {
        float4 v = s4[i];
        int row = (i * 4) >> 7;
        float sc = rowscale ? rowscale[row0 + row] : 1.0f;
        __bf16* o = Abf + i * 4;
        o[0] = f2bf(v.x * sc); o[1] = f2bf(v.y * sc);
        o[2] = f2bf(v.z * sc); o[3] = f2bf(v.w * sc);
    }
    if (tid == 0 && blockIdx.x + 1 < gridDim.x)
        __builtin_prefetch(A + (size_t)(row0 + 32) * HD, 0, 1);
    __syncthreads();

    const int wave = tid >> 5, lane = tid & 31;
    const int lm = lane & 15, lhi = lane >> 4;

    const v16bf* bf = (const v16bf*)wfrag;
    v16bf breg[4];
#pragma unroll
    for (int kt = 0; kt < 4; ++kt)
        breg[kt] = bf[(size_t)(kt * 8 + wave) * 32 + lane];

#pragma unroll
    for (int rt = 0; rt < 2; ++rt) {
        const int arow = rt * 16 + lm;
        v8f acc;
        if (accumulate) {
#pragma unroll
            for (int v = 0; v < 8; ++v)
                acc[v] = C[(size_t)(row0 + rt * 16 + v + 8 * lhi) * HD + wave * 16 + lm];
        } else {
            float bv = bias[wave * 16 + lm];
#pragma unroll
            for (int v = 0; v < 8; ++v) acc[v] = bv;
        }
#pragma unroll
        for (int kt = 0; kt < 4; ++kt) {
            const int kb = kt * 32 + 8 * lhi;
            v16bf a = cat8(*(const v8bf*)(Abf + arow * HD + kb),
                           *(const v8bf*)(Abf + arow * HD + kb + 16));
            acc = __builtin_amdgcn_wmma_f32_16x16x32_bf16(false, a, false, breg[kt],
                                                          (short)0, acc, false, false);
        }
#pragma unroll
        for (int v = 0; v < 8; ++v) {
            float r = acc[v];
            if (relu) r = r > 0.f ? r : 0.f;
            C[(size_t)(row0 + rt * 16 + v + 8 * lhi) * HD + wave * 16 + lm] = r;
        }
    }
}

// ---------------------------------------------------------------- launcher
extern "C" void kernel_launch(void* const* d_in, const int* in_sizes, int n_in,
                              void* d_out, int out_size, void* d_ws, size_t ws_size,
                              hipStream_t stream) {
    (void)in_sizes; (void)n_in; (void)out_size; (void)ws_size;

    const float* x_p  = (const float*)d_in[0];
    const float* x_g  = (const float*)d_in[1];
    const float* x_pa = (const float*)d_in[2];
    const float* w_in = (const float*)d_in[3];
    const float* b_in = (const float*)d_in[4];
    const float* wl   = (const float*)d_in[5];
    const float* bl   = (const float*)d_in[6];
    const float* wr   = (const float*)d_in[7];
    const int* ei_pp  = (const int*)d_in[8];
    const int* ei_pg  = (const int*)d_in[9];
    const int* ei_gp  = (const int*)d_in[10];
    const int* pa_src = (const int*)d_in[11];
    const int* pa_dst = (const int*)d_in[12];

    float* out_p  = (float*)d_out;
    float* out_g  = out_p + (size_t)NP * HD;
    float* out_pa = out_g + (size_t)NG * HD;

    // ---- carve workspace
    char* base = (char*)d_ws;
    size_t off = 0;
    auto carve = [&](size_t bytes) -> void* {
        off = (off + 255) & ~(size_t)255;
        void* p = base + off;
        off += bytes;
        return p;
    };
    float* h0[3] = { (float*)carve((size_t)NP * HD * 4),
                     (float*)carve((size_t)NG * HD * 4),
                     (float*)carve((size_t)NPA * HD * 4) };
    float* h1[3] = { (float*)carve((size_t)NP * HD * 4),
                     (float*)carve((size_t)NG * HD * 4),
                     (float*)carve((size_t)NPA * HD * 4) };
    float* agg = (float*)carve((size_t)NP * HD * 4);      // max dst-set size
    int*   cnt[4]; float* invc[4];
    const int ndst_r[4] = { NP, NG, NP, NPA };
    for (int r = 0; r < 4; ++r) cnt[r]  = (int*)carve((size_t)ndst_r[r] * 4);
    for (int r = 0; r < 4; ++r) invc[r] = (float*)carve((size_t)ndst_r[r] * 4);
    __bf16* winf = (__bf16*)carve((size_t)3 * (INF / 32) * 8 * 512 * 2);
    __bf16* wlf  = (__bf16*)carve((size_t)8 * 32 * 512 * 2);   // (l,r) x 32 frags
    __bf16* wrf  = (__bf16*)carve((size_t)6 * 32 * 512 * 2);   // (l,dt) x 32 frags
    float*  beff = (float*)carve((size_t)6 * HD * 4);

    auto zgrid = [](size_t n) -> unsigned {
        size_t b = (n + 255) / 256;
        return (unsigned)(b > 4096 ? 4096 : (b ? b : 1));
    };

    // ---- pack weights into WMMA fragments (bf16)
    for (int t = 0; t < 3; ++t)
        pack_frags<<<(INF / 32) * 8, 32, 0, stream>>>(
            w_in + (size_t)t * INF * HD, nullptr, 1.0f,
            winf + (size_t)t * (INF / 32) * 8 * 512);
    for (int l = 0; l < 2; ++l)
        for (int r = 0; r < 4; ++r) {
            float s = (r == 0 || r == 2) ? 0.5f : 1.0f;   // HeteroConv mean over relations
            pack_frags<<<32, 32, 0, stream>>>(
                wl + (size_t)(l * 4 + r) * HD * HD, nullptr, s,
                wlf + (size_t)(l * 4 + r) * 32 * 512);
        }
    for (int l = 0; l < 2; ++l) {
        pack_frags<<<32, 32, 0, stream>>>(wr + (size_t)(l * 4 + 0) * HD * HD,
                                          wr + (size_t)(l * 4 + 2) * HD * HD, 0.5f,
                                          wrf + (size_t)(l * 3 + 0) * 32 * 512);
        pack_frags<<<32, 32, 0, stream>>>(wr + (size_t)(l * 4 + 1) * HD * HD, nullptr, 1.0f,
                                          wrf + (size_t)(l * 3 + 1) * 32 * 512);
        pack_frags<<<32, 32, 0, stream>>>(wr + (size_t)(l * 4 + 3) * HD * HD, nullptr, 1.0f,
                                          wrf + (size_t)(l * 3 + 2) * 32 * 512);
    }
    beff_kernel<<<6, 128, 0, stream>>>(bl, beff);

    // ---- degrees -> 1/max(deg,1), per relation
    const int* src_r[4] = { ei_pp, ei_pg, ei_gp, pa_src };
    const int* dst_r[4] = { ei_pp + NEDGE, ei_pg + NEDGE, ei_gp + NEDGE, pa_dst };
    const int st_r[4] = { 0, 0, 1, 0 };
    const int dt_r[4] = { 0, 1, 0, 2 };
    for (int r = 0; r < 4; ++r) {
        zero_kernel<<<zgrid(ndst_r[r]), 256, 0, stream>>>((unsigned*)cnt[r], (size_t)ndst_r[r]);
        degree_kernel<<<(NEDGE + 255) / 256, 256, 0, stream>>>(dst_r[r], NEDGE, cnt[r]);
        inv_kernel<<<(ndst_r[r] + 255) / 256, 256, 0, stream>>>(cnt[r], invc[r], ndst_r[r]);
    }

    // ---- input projection + ReLU
    proj_wmma<<<NP / 32, 256, 0, stream>>>(
        x_p, winf + 0 * (size_t)(INF / 32) * 8 * 512, b_in + 0 * HD, h0[0]);
    proj_wmma<<<NG / 32, 256, 0, stream>>>(
        x_g, winf + 1 * (size_t)(INF / 32) * 8 * 512, b_in + 1 * HD, h0[1]);
    proj_wmma<<<NPA / 32, 256, 0, stream>>>(
        x_pa, winf + 2 * (size_t)(INF / 32) * 8 * 512, b_in + 2 * HD, h0[2]);

    // ---- 2 HeteroConv layers
    const int nnode[3] = { NP, NG, NPA };
    float* cur[3] = { h0[0], h0[1], h0[2] };
    for (int l = 0; l < 2; ++l) {
        float* nxt[3];
        if (l == 0) { nxt[0] = h1[0]; nxt[1] = h1[1]; nxt[2] = h1[2]; }
        else        { nxt[0] = out_p; nxt[1] = out_g; nxt[2] = out_pa; }

        // root term: nxt[dt] = cur[dt] @ wr_eff + b_eff   (initializes nxt)
        for (int dt = 0; dt < 3; ++dt)
            gemm128_wmma<<<nnode[dt] / 32, 256, 0, stream>>>(
                cur[dt], nullptr, wrf + (size_t)(l * 3 + dt) * 32 * 512,
                beff + (size_t)(l * 3 + dt) * HD, nxt[dt], 0, 0);

        // neighbor terms, one relation at a time (agg buffer reused).
        // Last accumulating GEMM per dst type in layer 0 fuses the ReLU:
        // dt0's last writer is r2, dt1's is r1, dt2's is r3.
        for (int r = 0; r < 4; ++r) {
            int nd = ndst_r[r];
            int relu = (l == 0 && r != 0) ? 1 : 0;
            zero_kernel<<<zgrid((size_t)nd * HD), 256, 0, stream>>>(
                (unsigned*)agg, (size_t)nd * HD);
            scatter_kernel<<<(unsigned)(((long)NEDGE * 32 + 255) / 256), 256, 0, stream>>>(
                cur[st_r[r]], src_r[r], dst_r[r], agg, NEDGE);
            gemm128_wmma<<<nd / 32, 256, 0, stream>>>(
                agg, invc[r], wlf + (size_t)(l * 4 + r) * 32 * 512, nullptr,
                nxt[dt_r[r]], 1, relu);
        }

        cur[0] = nxt[0]; cur[1] = nxt[1]; cur[2] = nxt[2];
    }
}
